// Transformer_13975823581963
// MI455X (gfx1250) — matrix operand order repaired and checked by reading, not compiled, not run on hardware
//
#include <hip/hip_runtime.h>
#include <hip/hip_bf16.h>

// ---------------------------------------------------------------------------
// Transformer forward for gfx1250 (MI455X).
//  - All GEMMs: V_WMMA_F32_16X16X32_BF16, fp32 accumulate.
//  - Weights pre-converted+transposed to bf16 [N][K] once per launch; weight
//    GEMM staging is raw 16B copies via GLOBAL_LOAD_ASYNC_TO_LDS_B128
//    (ASYNCcnt / s_wait_asynccnt), double-buffered LDS, 1 barrier per K-step.
//  - GEMM templated on B-operand kind so each instantiation has a
//    straight-line staging path (no per-iteration branching).
// ---------------------------------------------------------------------------

#define BATCH  4
#define SEQ    512
#define DMODEL 512
#define NHEAD  8
#define DKH    64
#define FFDIM  2048
#define NLAYER 6
#define VOCAB  32000

typedef __attribute__((ext_vector_type(16))) __bf16        v16bf;
typedef __attribute__((ext_vector_type(8)))  float         v8f;
typedef __attribute__((ext_vector_type(4)))  unsigned int  u32x4;

union FragAB { v16bf bf; u32x4 q[2]; };

__device__ __forceinline__ unsigned short f2bf(float f) {
  unsigned int u = __float_as_uint(f);
  u += 0x7FFFu + ((u >> 16) & 1u);          // round-to-nearest-even
  return (unsigned short)(u >> 16);
}

#if __has_builtin(__builtin_amdgcn_cvt_pk_bf16_f32)
typedef __attribute__((ext_vector_type(2))) __bf16 v2bf;
__device__ __forceinline__ unsigned int pack2(float a, float b) {
  union { v2bf v; unsigned int u; } r;
  r.v = __builtin_amdgcn_cvt_pk_bf16_f32(a, b);
  return r.u;
}
#else
__device__ __forceinline__ unsigned int pack2(float a, float b) {
  return (unsigned int)f2bf(a) | ((unsigned int)f2bf(b) << 16);
}
#endif

// Flat LDS pointers carry the LDS byte offset in addr[31:0] (ISA 10.2).
__device__ __forceinline__ unsigned int lds_off(const void* p) {
  return (unsigned int)(unsigned long long)p;
}
__device__ __forceinline__ void async_copy_b128(unsigned int dst_lds, const void* src) {
  asm volatile("global_load_async_to_lds_b128 %0, %1, off"
               :: "v"(dst_lds), "v"(src) : "memory");
}
__device__ __forceinline__ void wait_async0() {
  asm volatile("s_wait_asynccnt 0x0" ::: "memory");
}

// ---------------------------------------------------------------------------
// Weight convert + transpose: fp32 W[K][N] -> bf16 Wt[N][K]. gridDim.z = L.
// ---------------------------------------------------------------------------
__global__ void conv_transpose_bf16(const float* __restrict__ W,
                                    unsigned short* __restrict__ Wt,
                                    int K, int N) {
  __shared__ unsigned short t[32][33];
  const long long mat = (long long)blockIdx.z * K * N;
  const int n0 = blockIdx.x * 32, k0 = blockIdx.y * 32;
  const int tx = threadIdx.x, ty = threadIdx.y;
  #pragma unroll
  for (int j = 0; j < 4; ++j) {
    const int k = ty + j * 8;
    t[k][tx] = f2bf(W[mat + (long long)(k0 + k) * N + n0 + tx]);
  }
  __syncthreads();
  #pragma unroll
  for (int j = 0; j < 4; ++j) {
    const int n = ty + j * 8;
    Wt[mat + (long long)(n0 + n) * K + k0 + tx] = t[tx][n];
  }
}

// ---------------------------------------------------------------------------
// Batched GEMM: C = alpha * (A @ B) [+ bias] [relu]
//   A: fp32 [M][K] (converted to bf16 while staging)
//   BKIND 0 -> B fp32 [K][N]  (runtime V matrix; transpose-convert staging)
//   BKIND 1 -> B fp32 [N][K]  (runtime K matrix; convert staging)
//   BKIND 2 -> B bf16 [N][K]  (pre-converted weight; raw async 16B copies)
// Block: 256 threads / 8 waves; block tile 128x64; wave tile 32x32
// (4 WMMAs per K-step). M%128==0, N%64==0, K%32==0 for every GEMM here.
// Double-buffered LDS: one barrier per K-step.
// ---------------------------------------------------------------------------
template <int BKIND>
__launch_bounds__(256)
__global__ void gemm_bf16_wmma(const float* __restrict__ A, long long sAb, long long sAh, int lda,
                               const void* __restrict__ Bv, long long sBb, long long sBh, int ldb,
                               float*       __restrict__ C, long long sCb, long long sCh, int ldc,
                               int K, const float* __restrict__ bias, float alpha, int relu, int batH) {
  __shared__ unsigned short As[2][128][40];  // 80B rows, 16B aligned, pad vs conflicts
  __shared__ unsigned short Bs[2][64][40];   // B^T tile [n][k]

  const int tid  = threadIdx.x;
  const int lane = tid & 31, wid = tid >> 5;
  const int l16  = lane & 15, hi = lane >> 4;
  const int m0   = (wid & 3) * 32;           // wave rows m0..m0+31
  const int n0   = (wid >> 2) * 32;          // wave cols n0..n0+31

  const int bb = blockIdx.z / batH, hh = blockIdx.z % batH;
  A += bb * sAb + hh * sAh;
  C += bb * sCb + hh * sCh;
  const float*          Bf = (const float*)Bv + bb * sBb + hh * sBh;
  const unsigned short* Bh = (const unsigned short*)Bv + bb * sBb + hh * sBh;

  const long long rowA = (long long)blockIdx.y * 128;
  const long long colB = (long long)blockIdx.x * 64;

  const int ar  = tid >> 1, ac = (tid & 1) * 16;  // A stage: 128 rows x 32 k
  const int b0k = tid >> 3, b0n = (tid & 7) * 8;  // BKIND0: 32 k x 64 n
  const int b1n = tid >> 2, b1c = (tid & 3) * 8;  // BKIND1/2: 64 n x 32 k

  auto stageA = [&](int buf, int k0) {
    const float* sp = A + (rowA + ar) * lda + (k0 + ac);
    __builtin_prefetch(sp + 32, 0, 1);
    u32x4 p0, p1;
    p0[0] = pack2(sp[0],  sp[1]);  p0[1] = pack2(sp[2],  sp[3]);
    p0[2] = pack2(sp[4],  sp[5]);  p0[3] = pack2(sp[6],  sp[7]);
    p1[0] = pack2(sp[8],  sp[9]);  p1[1] = pack2(sp[10], sp[11]);
    p1[2] = pack2(sp[12], sp[13]); p1[3] = pack2(sp[14], sp[15]);
    *(u32x4*)&As[buf][ar][ac]     = p0;
    *(u32x4*)&As[buf][ar][ac + 8] = p1;
  };
  auto stageB = [&](int buf, int k0) {
    if constexpr (BKIND == 2) {        // bf16 [N][K]: raw async copy to LDS
      const unsigned short* sp = Bh + (colB + b1n) * (long long)ldb + k0 + b1c;
      async_copy_b128(lds_off(&Bs[buf][b1n][b1c]), sp);
    } else if constexpr (BKIND == 1) { // fp32 [N][K]: convert, b128 store
      const float* sp = Bf + (colB + b1n) * (long long)ldb + k0 + b1c;
      u32x4 p;
      p[0] = pack2(sp[0], sp[1]); p[1] = pack2(sp[2], sp[3]);
      p[2] = pack2(sp[4], sp[5]); p[3] = pack2(sp[6], sp[7]);
      *(u32x4*)&Bs[buf][b1n][b1c] = p;
    } else {                           // fp32 [K][N]: coalesced read, transposed store
      const float* sp = Bf + (long long)(k0 + b0k) * ldb + colB + b0n;
      #pragma unroll
      for (int j = 0; j < 8; ++j) Bs[buf][b0n + j][b0k] = f2bf(sp[j]);
    }
  };

  v8f acc[2][2] = {};

  stageA(0, 0);
  stageB(0, 0);
  if constexpr (BKIND == 2) wait_async0();
  __syncthreads();

  int cur = 0;
  for (int k0 = 0; k0 < K; k0 += 32) {
    const int nxt = cur ^ 1;
    const bool more = (k0 + 32) < K;
    if (more) { stageA(nxt, k0 + 32); stageB(nxt, k0 + 32); }

    FragAB a[2], b[2];
    a[0].q[0] = *(const u32x4*)&As[cur][m0 + l16][hi * 8];
    a[0].q[1] = *(const u32x4*)&As[cur][m0 + l16][hi * 8 + 16];
    a[1].q[0] = *(const u32x4*)&As[cur][m0 + 16 + l16][hi * 8];
    a[1].q[1] = *(const u32x4*)&As[cur][m0 + 16 + l16][hi * 8 + 16];
    const u32x4* pb0 = (const u32x4*)&Bs[cur][n0 + l16][hi * 8];
    b[0].q[0] = pb0[0]; b[0].q[1] = pb0[2];
    const u32x4* pb1 = (const u32x4*)&Bs[cur][n0 + 16 + l16][hi * 8];
    b[1].q[0] = pb1[0]; b[1].q[1] = pb1[2];

    acc[0][0] = __builtin_amdgcn_wmma_f32_16x16x32_bf16(false, a[0].bf, false, b[0].bf,
                                                        (short)0, acc[0][0], false, false);
    acc[0][1] = __builtin_amdgcn_wmma_f32_16x16x32_bf16(false, a[0].bf, false, b[1].bf,
                                                        (short)0, acc[0][1], false, false);
    acc[1][0] = __builtin_amdgcn_wmma_f32_16x16x32_bf16(false, a[1].bf, false, b[0].bf,
                                                        (short)0, acc[1][0], false, false);
    acc[1][1] = __builtin_amdgcn_wmma_f32_16x16x32_bf16(false, a[1].bf, false, b[1].bf,
                                                        (short)0, acc[1][1], false, false);
    cur = nxt;
    if (more) {
      if constexpr (BKIND == 2) wait_async0();
      __syncthreads();
    }
  }

  // C/D layout: VGPR j -> row (j + 8*hi), col l16.
  const long long gn = colB + n0 + l16;
  #pragma unroll
  for (int s = 0; s < 2; ++s) {
    const long long gm = rowA + m0 + s * 16 + hi * 8;
    #pragma unroll
    for (int j = 0; j < 8; ++j) {
      float v0 = acc[s][0][j] * alpha;
      float v1 = acc[s][1][j] * alpha;
      if (bias) { v0 += bias[gn]; v1 += bias[gn + 16]; }
      if (relu) { v0 = fmaxf(v0, 0.f); v1 = fmaxf(v1, 0.f); }
      C[(gm + j) * ldc + gn]      = v0;
      C[(gm + j) * ldc + gn + 16] = v1;
    }
  }
}

// ---------------------------------------------------------------------------
// Row softmax: one wave per row (8 rows / 256-thread block), wave32 shuffles.
// ---------------------------------------------------------------------------
__global__ void softmax_rows(float* __restrict__ sc, int cols) {
  const int lane = threadIdx.x & 31;
  const long long row = (long long)blockIdx.x * 8 + (threadIdx.x >> 5);
  float* r = sc + row * cols;
  float m = -3.402823e38f;
  for (int i = lane; i < cols; i += 32) m = fmaxf(m, r[i]);
  for (int o = 16; o > 0; o >>= 1) m = fmaxf(m, __shfl_xor(m, o, 32));
  float s = 0.f;
  for (int i = lane; i < cols; i += 32) { float e = __expf(r[i] - m); r[i] = e; s += e; }
  for (int o = 16; o > 0; o >>= 1) s += __shfl_xor(s, o, 32);
  const float inv = 1.0f / s;
  for (int i = lane; i < cols; i += 32) r[i] *= inv;
}

// ---------------------------------------------------------------------------
// Fused residual + LayerNorm over D=512: x = LN(x + res)*g + b, in place.
// ---------------------------------------------------------------------------
__global__ void add_ln(float* __restrict__ x, const float* __restrict__ res,
                       const float* __restrict__ g, const float* __restrict__ bta) {
  __shared__ float rs[4], rs2[4];
  const int tid = threadIdx.x;
  const long long row = blockIdx.x;
  float v[4]; float s = 0.f, s2 = 0.f;
  #pragma unroll
  for (int j = 0; j < 4; ++j) {
    const int idx = tid + j * 128;
    const float t = x[row * DMODEL + idx] + res[row * DMODEL + idx];
    v[j] = t; s += t; s2 += t * t;
  }
  for (int o = 16; o > 0; o >>= 1) { s += __shfl_xor(s, o, 32); s2 += __shfl_xor(s2, o, 32); }
  const int lane = tid & 31, wid = tid >> 5;
  if (lane == 0) { rs[wid] = s; rs2[wid] = s2; }
  __syncthreads();
  s  = rs[0] + rs[1] + rs[2] + rs[3];
  s2 = rs2[0] + rs2[1] + rs2[2] + rs2[3];
  const float mean = s * (1.f / DMODEL);
  const float var  = s2 * (1.f / DMODEL) - mean * mean;
  const float rstd = rsqrtf(var + 1e-5f);
  #pragma unroll
  for (int j = 0; j < 4; ++j) {
    const int idx = tid + j * 128;
    x[row * DMODEL + idx] = (v[j] - mean) * rstd * g[idx] + bta[idx];
  }
}

// ---------------------------------------------------------------------------
// Token embedding * sqrt(D) + sinusoidal positional encoding (computed here).
// ---------------------------------------------------------------------------
__global__ void embed_pe(const int* __restrict__ tok, const float* __restrict__ emb,
                         float* __restrict__ out) {
  const long long row = blockIdx.x;               // b*SEQ + s
  const int s = (int)(row % SEQ);
  const long long t = tok[row];
  const float scale = 22.62741699796952f;         // sqrt(512)
  for (int d = threadIdx.x; d < DMODEL; d += blockDim.x) {
    const float expo = (float)(d & ~1) * (-9.210340371976184f / (float)DMODEL);
    const float ang  = (float)s * __expf(expo);
    const float pe   = (d & 1) ? __cosf(ang) : __sinf(ang);
    out[row * DMODEL + d] = emb[t * DMODEL + d] * scale + pe;
  }
}

// ---------------------------------------------------------------------------
// Host orchestration. Input order = recursive pytree flatten (sorted keys):
//   Wout, bout, dec.{...}, enc.{...}, src, src_emb, tgt, tgt_emb.
// ---------------------------------------------------------------------------
extern "C" void kernel_launch(void* const* d_in, const int* in_sizes, int n_in,
                              void* d_out, int out_size, void* d_ws, size_t ws_size,
                              hipStream_t stream) {
  (void)in_sizes; (void)n_in; (void)out_size; (void)ws_size;
  const float* Wout  = (const float*)d_in[0];
  const float* bout  = (const float*)d_in[1];
  const float* dW1   = (const float*)d_in[2];
  const float* dW2   = (const float*)d_in[3];
  const float* db1   = (const float*)d_in[4];
  const float* db2   = (const float*)d_in[5];
  const float* dbe1  = (const float*)d_in[6];
  const float* dbe2  = (const float*)d_in[7];
  const float* dbe3  = (const float*)d_in[8];
  const float* dcaWk = (const float*)d_in[9];
  const float* dcaWo = (const float*)d_in[10];
  const float* dcaWq = (const float*)d_in[11];
  const float* dcaWv = (const float*)d_in[12];
  const float* dcabk = (const float*)d_in[13];
  const float* dcabo = (const float*)d_in[14];
  const float* dcabq = (const float*)d_in[15];
  const float* dcabv = (const float*)d_in[16];
  const float* dg1   = (const float*)d_in[17];
  const float* dg2   = (const float*)d_in[18];
  const float* dg3   = (const float*)d_in[19];
  const float* dsaWk = (const float*)d_in[20];
  const float* dsaWo = (const float*)d_in[21];
  const float* dsaWq = (const float*)d_in[22];
  const float* dsaWv = (const float*)d_in[23];
  const float* dsabk = (const float*)d_in[24];
  const float* dsabo = (const float*)d_in[25];
  const float* dsabq = (const float*)d_in[26];
  const float* dsabv = (const float*)d_in[27];
  const float* eW1   = (const float*)d_in[28];
  const float* eW2   = (const float*)d_in[29];
  const float* eb1   = (const float*)d_in[30];
  const float* eb2   = (const float*)d_in[31];
  const float* ebe1  = (const float*)d_in[32];
  const float* ebe2  = (const float*)d_in[33];
  const float* eg1   = (const float*)d_in[34];
  const float* eg2   = (const float*)d_in[35];
  const float* esaWk = (const float*)d_in[36];
  const float* esaWo = (const float*)d_in[37];
  const float* esaWq = (const float*)d_in[38];
  const float* esaWv = (const float*)d_in[39];
  const float* esabk = (const float*)d_in[40];
  const float* esabo = (const float*)d_in[41];
  const float* esabq = (const float*)d_in[42];
  const float* esabv = (const float*)d_in[43];
  const int*   src     = (const int*)d_in[44];
  const float* src_emb = (const float*)d_in[45];
  const int*   tgt     = (const int*)d_in[46];
  const float* tgt_emb = (const float*)d_in[47];

  char* ws = (char*)d_ws;
  size_t off = 0;
  auto alloc = [&](size_t bytes) -> void* {
    void* p = ws + off;
    off = (off + bytes + 255) & ~(size_t)255;
    return p;
  };
  auto halloc = [&](size_t elems) -> unsigned short* {
    return (unsigned short*)alloc(elems * sizeof(unsigned short));
  };

  const size_t actB = (size_t)BATCH * SEQ * DMODEL * sizeof(float);
  float* x   = (float*)alloc(actB);   // encoder state / enc_out
  float* y   = (float*)alloc(actB);   // decoder state
  float* tmp = (float*)alloc(actB);
  float* Qb  = (float*)alloc(actB);
  float* Kb  = (float*)alloc(actB);
  float* Vb  = (float*)alloc(actB);
  float* Cx  = (float*)alloc(actB);
  float* hid = (float*)alloc((size_t)BATCH * SEQ * FFDIM * sizeof(float));
  float* Sc  = (float*)alloc((size_t)BATCH * NHEAD * SEQ * SEQ * sizeof(float));

  const size_t DD = (size_t)DMODEL * DMODEL;
  const size_t DF = (size_t)DMODEL * FFDIM;
  unsigned short* tesaWq = halloc(NLAYER * DD);
  unsigned short* tesaWk = halloc(NLAYER * DD);
  unsigned short* tesaWv = halloc(NLAYER * DD);
  unsigned short* tesaWo = halloc(NLAYER * DD);
  unsigned short* teW1   = halloc(NLAYER * DF);
  unsigned short* teW2   = halloc(NLAYER * DF);
  unsigned short* tdsaWq = halloc(NLAYER * DD);
  unsigned short* tdsaWk = halloc(NLAYER * DD);
  unsigned short* tdsaWv = halloc(NLAYER * DD);
  unsigned short* tdsaWo = halloc(NLAYER * DD);
  unsigned short* tdcaWq = halloc(NLAYER * DD);
  unsigned short* tdcaWk = halloc(NLAYER * DD);
  unsigned short* tdcaWv = halloc(NLAYER * DD);
  unsigned short* tdcaWo = halloc(NLAYER * DD);
  unsigned short* tdW1   = halloc(NLAYER * DF);
  unsigned short* tdW2   = halloc(NLAYER * DF);
  unsigned short* tWout  = halloc((size_t)DMODEL * VOCAB);

  auto wconv = [&](const float* W, unsigned short* Wt, int K, int N, int L) {
    conv_transpose_bf16<<<dim3(N / 32, K / 32, L), dim3(32, 8, 1), 0, stream>>>(W, Wt, K, N);
  };
  wconv(esaWq, tesaWq, DMODEL, DMODEL, NLAYER);
  wconv(esaWk, tesaWk, DMODEL, DMODEL, NLAYER);
  wconv(esaWv, tesaWv, DMODEL, DMODEL, NLAYER);
  wconv(esaWo, tesaWo, DMODEL, DMODEL, NLAYER);
  wconv(eW1,   teW1,   DMODEL, FFDIM,  NLAYER);
  wconv(eW2,   teW2,   FFDIM,  DMODEL, NLAYER);
  wconv(dsaWq, tdsaWq, DMODEL, DMODEL, NLAYER);
  wconv(dsaWk, tdsaWk, DMODEL, DMODEL, NLAYER);
  wconv(dsaWv, tdsaWv, DMODEL, DMODEL, NLAYER);
  wconv(dsaWo, tdsaWo, DMODEL, DMODEL, NLAYER);
  wconv(dcaWq, tdcaWq, DMODEL, DMODEL, NLAYER);
  wconv(dcaWk, tdcaWk, DMODEL, DMODEL, NLAYER);
  wconv(dcaWv, tdcaWv, DMODEL, DMODEL, NLAYER);
  wconv(dcaWo, tdcaWo, DMODEL, DMODEL, NLAYER);
  wconv(dW1,   tdW1,   DMODEL, FFDIM,  NLAYER);
  wconv(dW2,   tdW2,   FFDIM,  DMODEL, NLAYER);
  wconv(Wout,  tWout,  DMODEL, VOCAB,  1);

  auto gemm = [&](const float* A, long long sAb, long long sAh, int lda,
                  const void* Bp, long long sBb, long long sBh, int ldb, int bkind,
                  float* Cp, long long sCb, long long sCh, int ldc,
                  int M, int N, int K, const float* bias, float alpha, int relu,
                  int batB, int batH) {
    dim3 grid(N / 64, M / 128, batB * batH);
    if (bkind == 2)
      gemm_bf16_wmma<2><<<grid, dim3(256, 1, 1), 0, stream>>>(
          A, sAb, sAh, lda, Bp, sBb, sBh, ldb, Cp, sCb, sCh, ldc, K, bias, alpha, relu, batH);
    else if (bkind == 1)
      gemm_bf16_wmma<1><<<grid, dim3(256, 1, 1), 0, stream>>>(
          A, sAb, sAh, lda, Bp, sBb, sBh, ldb, Cp, sCb, sCh, ldc, K, bias, alpha, relu, batH);
    else
      gemm_bf16_wmma<0><<<grid, dim3(256, 1, 1), 0, stream>>>(
          A, sAb, sAh, lda, Bp, sBb, sBh, ldb, Cp, sCb, sCh, ldc, K, bias, alpha, relu, batH);
  };

  auto mha = [&](const float* qin, const float* kvin, int Sq, int Sk,
                 const unsigned short* Wq, const float* bq,
                 const unsigned short* Wk, const float* bk,
                 const unsigned short* Wv, const float* bv,
                 const unsigned short* Wo, const float* bo,
                 float* outp) {
    gemm(qin,  0, 0, DMODEL, Wq, 0, 0, DMODEL, 2, Qb, 0, 0, DMODEL,
         BATCH * Sq, DMODEL, DMODEL, bq, 1.f, 0, 1, 1);
    gemm(kvin, 0, 0, DMODEL, Wk, 0, 0, DMODEL, 2, Kb, 0, 0, DMODEL,
         BATCH * Sk, DMODEL, DMODEL, bk, 1.f, 0, 1, 1);
    gemm(kvin, 0, 0, DMODEL, Wv, 0, 0, DMODEL, 2, Vb, 0, 0, DMODEL,
         BATCH * Sk, DMODEL, DMODEL, bv, 1.f, 0, 1, 1);
    // scores[b,h] = Q @ K^T / sqrt(DK)  (K matrix fp32 [N][K] -> BKIND 1)
    gemm(Qb, (long long)Sq * DMODEL, DKH, DMODEL,
         Kb, (long long)Sk * DMODEL, DKH, DMODEL, 1,
         Sc, (long long)NHEAD * Sq * Sk, (long long)Sq * Sk, Sk,
         Sq, Sk, DKH, nullptr, 0.125f, 0, BATCH, NHEAD);
    softmax_rows<<<dim3((BATCH * NHEAD * Sq) / 8), dim3(256), 0, stream>>>(Sc, Sk);
    // ctx[b,h] = attn @ V  (V fp32 [K][N] -> BKIND 0)
    gemm(Sc, (long long)NHEAD * Sq * Sk, (long long)Sq * Sk, Sk,
         Vb, (long long)Sk * DMODEL, DKH, DMODEL, 0,
         Cx, (long long)Sq * DMODEL, DKH, DMODEL,
         Sq, DKH, Sk, nullptr, 1.f, 0, BATCH, NHEAD);
    gemm(Cx, 0, 0, DMODEL, Wo, 0, 0, DMODEL, 2, outp, 0, 0, DMODEL,
         BATCH * Sq, DMODEL, DMODEL, bo, 1.f, 0, 1, 1);
  };

  // ---------------- Encoder ----------------
  embed_pe<<<dim3(BATCH * SEQ), dim3(256), 0, stream>>>(src, src_emb, x);
  for (int i = 0; i < NLAYER; ++i) {
    const size_t wo = (size_t)i * DD, vo = (size_t)i * DMODEL;
    mha(x, x, SEQ, SEQ, tesaWq + wo, esabq + vo, tesaWk + wo, esabk + vo,
        tesaWv + wo, esabv + vo, tesaWo + wo, esabo + vo, tmp);
    add_ln<<<dim3(BATCH * SEQ), dim3(128), 0, stream>>>(x, tmp, eg1 + vo, ebe1 + vo);
    gemm(x, 0, 0, DMODEL, teW1 + (size_t)i * DF, 0, 0, DMODEL, 2,
         hid, 0, 0, FFDIM, BATCH * SEQ, FFDIM, DMODEL,
         eb1 + (size_t)i * FFDIM, 1.f, 1, 1, 1);
    gemm(hid, 0, 0, FFDIM, teW2 + (size_t)i * DF, 0, 0, FFDIM, 2,
         tmp, 0, 0, DMODEL, BATCH * SEQ, DMODEL, FFDIM, eb2 + vo, 1.f, 0, 1, 1);
    add_ln<<<dim3(BATCH * SEQ), dim3(128), 0, stream>>>(x, tmp, eg2 + vo, ebe2 + vo);
  }

  // ---------------- Decoder ----------------
  embed_pe<<<dim3(BATCH * SEQ), dim3(256), 0, stream>>>(tgt, tgt_emb, y);
  for (int i = 0; i < NLAYER; ++i) {
    const size_t wo = (size_t)i * DD, vo = (size_t)i * DMODEL;
    mha(y, y, SEQ, SEQ, tdsaWq + wo, dsabq + vo, tdsaWk + wo, dsabk + vo,
        tdsaWv + wo, dsabv + vo, tdsaWo + wo, dsabo + vo, tmp);
    add_ln<<<dim3(BATCH * SEQ), dim3(128), 0, stream>>>(y, tmp, dg1 + vo, dbe1 + vo);
    mha(y, x, SEQ, SEQ, tdcaWq + wo, dcabq + vo, tdcaWk + wo, dcabk + vo,
        tdcaWv + wo, dcabv + vo, tdcaWo + wo, dcabo + vo, tmp);
    add_ln<<<dim3(BATCH * SEQ), dim3(128), 0, stream>>>(y, tmp, dg2 + vo, dbe2 + vo);
    gemm(y, 0, 0, DMODEL, tdW1 + (size_t)i * DF, 0, 0, DMODEL, 2,
         hid, 0, 0, FFDIM, BATCH * SEQ, FFDIM, DMODEL,
         db1 + (size_t)i * FFDIM, 1.f, 1, 1, 1);
    gemm(hid, 0, 0, FFDIM, tdW2 + (size_t)i * DF, 0, 0, FFDIM, 2,
         tmp, 0, 0, DMODEL, BATCH * SEQ, DMODEL, FFDIM, db2 + vo, 1.f, 0, 1, 1);
    add_ln<<<dim3(BATCH * SEQ), dim3(128), 0, stream>>>(y, tmp, dg3 + vo, dbe3 + vo);
  }

  // ---------------- Output projection ----------------
  gemm(y, 0, 0, DMODEL, tWout, 0, 0, DMODEL, 2, (float*)d_out, 0, 0, VOCAB,
       BATCH * SEQ, VOCAB, DMODEL, bout, 1.f, 0, 1, 1);
}
